// GroupConv_6133213299287
// MI455X (gfx1250) — compile-verified, matrix operand order checked by reading, not run on hardware
//
#include <hip/hip_runtime.h>

// ---------------------------------------------------------------------------
// GroupConv (cyclic group, G=24) as 24 rolled GEMMs, f16 WMMA w/ f32 accum.
//   out[b,h,e] = sum_{k,i} X[b,(h+k)%G,i] * W[k,i,e] + bias[e]
// 128x128 WG tile, 64x32 wave tile (4x2 WMMA accumulators), double-buffered
// LDS staged with CDNA5 async global->LDS loads when available.
// ---------------------------------------------------------------------------

typedef __attribute__((ext_vector_type(4)))  _Float16 v4h;
typedef __attribute__((ext_vector_type(8)))  _Float16 v8h;
typedef __attribute__((ext_vector_type(16))) _Float16 v16h;
typedef __attribute__((ext_vector_type(4)))  float    v4f;
typedef __attribute__((ext_vector_type(8)))  float    v8f;
typedef __attribute__((ext_vector_type(4)))  int      v4i;

#define B_ 256
#define G_ 24
#define I_ 512
#define E_ 512

#define TM  128     // M tile (rows of B_)
#define TN  128     // N tile (cols of E_)
#define TK  32      // K chunk (== WMMA K)
#define LDH 40      // padded LDS row stride in halves (32 + 8)

#if defined(__HIP_DEVICE_COMPILE__) && \
    __has_builtin(__builtin_amdgcn_global_load_async_to_lds_b128)
#define HAVE_ASYNC_LDS 1
typedef __attribute__((address_space(1))) v4i* g4_ptr;   // global int4*
typedef __attribute__((address_space(3))) v4i* l4_ptr;   // LDS int4*
#else
#define HAVE_ASYNC_LDS 0
#endif

__device__ __forceinline__ void async_wait_all() {
#if HAVE_ASYNC_LDS
#if __has_builtin(__builtin_amdgcn_s_wait_asynccnt)
  __builtin_amdgcn_s_wait_asynccnt(0);
#else
  asm volatile("s_wait_asynccnt 0" ::: "memory");
#endif
#endif
}

// ---------------- prep: f32 -> f16 convert of input ------------------------
__global__ __launch_bounds__(256) void cvt_input_kernel(
    const float* __restrict__ x, _Float16* __restrict__ y) {
  int idx = (blockIdx.x * 256 + threadIdx.x) * 4;
  v4f v = *(const v4f*)(x + idx);
  *(v4h*)(y + idx) = __builtin_convertvector(v, v4h);
}

// ------- prep: weights [k][i][e] f32 -> WT [k][e][i] f16 (transpose) -------
__global__ __launch_bounds__(256) void cvt_weights_kernel(
    const float* __restrict__ w, _Float16* __restrict__ wt) {
  __shared__ float tile[32][33];
  const int k  = blockIdx.z;
  const int e0 = blockIdx.x * 32;
  const int i0 = blockIdx.y * 32;
  const int tx = threadIdx.x;          // 0..31
  const int ty = threadIdx.y;          // 0..7
#pragma unroll
  for (int r = ty; r < 32; r += 8)
    tile[r][tx] = w[((size_t)k * I_ + (i0 + r)) * E_ + e0 + tx];
  __syncthreads();
#pragma unroll
  for (int r = ty; r < 32; r += 8)
    wt[((size_t)k * E_ + (e0 + r)) * I_ + i0 + tx] = (_Float16)tile[tx][r];
}

// ---------------------------- main WMMA GEMM -------------------------------
__global__ __launch_bounds__(256) void gconv_wmma_kernel(
    const _Float16* __restrict__ X,    // [B][G][I]  f16
    const _Float16* __restrict__ WT,   // [G(k)][E][I] f16 (K-contiguous rows)
    const float*    __restrict__ bias, // [E]
    float*          __restrict__ out)  // [B][G(h)][E]
{
  __shared__ __align__(16) _Float16 As[2][TM * LDH];
  __shared__ __align__(16) _Float16 Bs[2][TN * LDH];

  const int h      = blockIdx.z;
  const int m_base = blockIdx.y * TM;
  const int n_base = blockIdx.x * TN;

  const int t      = threadIdx.x;
  const int lane   = t & 31;
  const int wave   = t >> 5;           // 0..7
  const int wave_m = wave >> 2;        // 0..1  -> rows wave_m*64
  const int wave_n = wave & 3;         // 0..3  -> cols wave_n*32

  // staging: 256 threads, A and B each 128 rows x 32 halves (2x b128/thread)
  const int srow = t >> 1;             // 0..127
  const int scol = (t & 1) * 16;       // halves

  // fragment addressing per ISA 16-bit A/B layout
  const int fr = lane & 15;            // row (A) / col (B) within 16
  const int k0 = (lane >> 4) * 8;      // per-lane runs [k0,k0+8) and [k0+16,k0+24)

  v8f acc[4][2] = {};

  const int NCHUNK = (G_ * I_) / TK;   // 384

  auto stage = [&](int chunk, int buf) {
    const int kg = chunk >> 4;                 // which group element k
    const int io = (chunk & 15) * TK;          // i offset inside [0,512)
    int g = h + kg; if (g >= G_) g -= G_;
    const _Float16* sa = X  + (((size_t)(m_base + srow)) * G_ + g) * I_ + io + scol;
    const _Float16* sb = WT + (((size_t)kg) * E_ + (n_base + srow)) * I_ + io + scol;
    _Float16* da = &As[buf][srow * LDH + scol];
    _Float16* db = &Bs[buf][srow * LDH + scol];
#if HAVE_ASYNC_LDS
    // global ptrs are identity-mapped in generic AS -> integer cast is safe
    // (and drops const); LDS needs a true addrspacecast.
    __builtin_amdgcn_global_load_async_to_lds_b128((g4_ptr)(uintptr_t)sa,
                                                   (l4_ptr)da, 0, 0);
    __builtin_amdgcn_global_load_async_to_lds_b128((g4_ptr)(uintptr_t)(sa + 8),
                                                   (l4_ptr)(da + 8), 0, 0);
    __builtin_amdgcn_global_load_async_to_lds_b128((g4_ptr)(uintptr_t)sb,
                                                   (l4_ptr)db, 0, 0);
    __builtin_amdgcn_global_load_async_to_lds_b128((g4_ptr)(uintptr_t)(sb + 8),
                                                   (l4_ptr)(db + 8), 0, 0);
#else
    *(uint4*)da       = *(const uint4*)sa;
    *(uint4*)(da + 8) = *(const uint4*)(sa + 8);
    *(uint4*)db       = *(const uint4*)sb;
    *(uint4*)(db + 8) = *(const uint4*)(sb + 8);
#endif
  };

  auto compute = [&](int buf) {
    v16h a[4], b[2];
#pragma unroll
    for (int mt = 0; mt < 4; ++mt) {
      const int r = wave_m * 64 + mt * 16 + fr;
      v8h lo = *(const v8h*)&As[buf][r * LDH + k0];
      v8h hi = *(const v8h*)&As[buf][r * LDH + k0 + 16];
      a[mt] = __builtin_shufflevector(lo, hi, 0,1,2,3,4,5,6,7,8,9,10,11,12,13,14,15);
    }
#pragma unroll
    for (int nt = 0; nt < 2; ++nt) {
      const int c = wave_n * 32 + nt * 16 + fr;
      v8h lo = *(const v8h*)&Bs[buf][c * LDH + k0];
      v8h hi = *(const v8h*)&Bs[buf][c * LDH + k0 + 16];
      b[nt] = __builtin_shufflevector(lo, hi, 0,1,2,3,4,5,6,7,8,9,10,11,12,13,14,15);
    }
#pragma unroll
    for (int mt = 0; mt < 4; ++mt)
#pragma unroll
      for (int nt = 0; nt < 2; ++nt)
        acc[mt][nt] = __builtin_amdgcn_wmma_f32_16x16x32_f16(
            false, a[mt], false, b[nt], (short)0, acc[mt][nt], false, false);
  };

  // double-buffered pipeline: load c+1 async while computing c
  stage(0, 0);
  async_wait_all();
  __syncthreads();
  for (int c = 0; c < NCHUNK; ++c) {
    const int cur = c & 1;
    if (c + 1 < NCHUNK) stage(c + 1, cur ^ 1);
    compute(cur);
    async_wait_all();
    __syncthreads();
  }

  // epilogue: C/D layout — VGPR j: M = j + 8*(lane>>4), N = lane&15
  const int eh = lane & 15;
  const int mh = (lane >> 4) * 8;
#pragma unroll
  for (int mt = 0; mt < 4; ++mt) {
#pragma unroll
    for (int nt = 0; nt < 2; ++nt) {
      const int e    = n_base + wave_n * 32 + nt * 16 + eh;
      const int mrow = m_base + wave_m * 64 + mt * 16 + mh;
      const float be = bias[e];
#pragma unroll
      for (int j = 0; j < 8; ++j)
        out[((size_t)(mrow + j)) * (G_ * E_) + (size_t)h * E_ + e] = acc[mt][nt][j] + be;
    }
  }
}

// ---------------------------------------------------------------------------
extern "C" void kernel_launch(void* const* d_in, const int* in_sizes, int n_in,
                              void* d_out, int out_size, void* d_ws, size_t ws_size,
                              hipStream_t stream) {
  (void)in_sizes; (void)n_in; (void)out_size; (void)ws_size;
  const float* input   = (const float*)d_in[0];
  const float* weights = (const float*)d_in[1];
  const float* bias    = (const float*)d_in[2];
  // d_in[3] (inv_indices) encodes (g-h)%G — derived analytically in-kernel.

  _Float16* X16  = (_Float16*)d_ws;                                     // 6.29 MB
  _Float16* WT16 = (_Float16*)((char*)d_ws + (size_t)B_ * G_ * I_ * 2); // 12.58 MB

  cvt_input_kernel<<<(B_ * G_ * I_) / 1024, 256, 0, stream>>>(input, X16);
  cvt_weights_kernel<<<dim3(E_ / 32, I_ / 32, G_), dim3(32, 8), 0, stream>>>(weights, WT16);
  gconv_wmma_kernel<<<dim3(E_ / TN, B_ / TM, G_), 256, 0, stream>>>(
      X16, WT16, bias, (float*)d_out);
}
